// Encoder_87634512708306
// MI455X (gfx1250) — matrix operand order repaired and checked by reading, not compiled, not run on hardware
//
#include <hip/hip_runtime.h>
#include <hip/hip_bf16.h>

// ---------------------------------------------------------------------------
// Transformer encoder forward for MI455X (gfx1250, wave32, WMMA).
// B=8 S=512 D=1024 H=16 DQ=DV=64 DI=4096 L=6
// All GEMMs + attention einsums use v_wmma_f32_16x16x32_f16.
// GEMM: 128x256 block tile, 64x64 wave tile (16 WMMA / k-step / wave),
//       global_prefetch of the next k-tile.
// ---------------------------------------------------------------------------

typedef __attribute__((ext_vector_type(16))) _Float16 v16h;
typedef __attribute__((ext_vector_type(8)))  _Float16 h8;
typedef __attribute__((ext_vector_type(8)))  float    v8f;

#define BB 8
#define SS 512
#define DD 1024
#define HH 16
#define DHEAD 64
#define DI 4096
#define NL 6
#define NTOK (BB*SS)   // 4096

__device__ __forceinline__ v8f wmma16(v16h a, v16h b, v8f c) {
  // D = A(16x32 f16) * B(32x16 f16) + C(16x16 f32)
  return __builtin_amdgcn_wmma_f32_16x16x32_f16(
      /*neg_a=*/false, a, /*neg_b=*/false, b,
      /*c_mod=*/(short)0, c, /*reuse_a=*/false, /*reuse_b=*/false);
}

// A fragment (16x32 f16, row-major tile with leading dim ld):
// lanes 0-15: row m, VGPR0-3 = K0..7, VGPR4-7 = K16..23
// lanes 16-31: row m, VGPR0-3 = K8..15, VGPR4-7 = K24..31
__device__ __forceinline__ v16h ld_frag_a(const _Float16* p, int ld, int m,
                                          int k0, int half) {
  h8 lo = *(const h8*)(p + m * ld + k0 + half * 8);
  h8 hi = *(const h8*)(p + m * ld + k0 + 16 + half * 8);
  return __builtin_shufflevector(lo, hi, 0,1,2,3,4,5,6,7,8,9,10,11,12,13,14,15);
}

// B fragment (32x16 f16) from a K-contiguous (transposed) tile BT[n][k]:
// lanes 0-15: col n, K0..7,K16..23 ; lanes 16-31: col n, K8..15,K24..31
__device__ __forceinline__ v16h ld_frag_b(const _Float16* p, int ld, int n,
                                          int k0, int half) {
  h8 lo = *(const h8*)(p + n * ld + k0 + half * 8);
  h8 hi = *(const h8*)(p + n * ld + k0 + 16 + half * 8);
  return __builtin_shufflevector(lo, hi, 0,1,2,3,4,5,6,7,8,9,10,11,12,13,14,15);
}

// ---------------------------------------------------------------------------
// Embedding + sinusoidal positional encoding -> f32 x[NTOK][D]
// ---------------------------------------------------------------------------
__global__ __launch_bounds__(256) void embed_k(const int* __restrict__ tokens,
                                               const float* __restrict__ emb,
                                               float* __restrict__ x) {
  const int bs = blockIdx.x;         // 0..4095
  const int s  = bs & (SS - 1);
  const int tok = tokens[bs];
  for (int d = threadIdx.x; d < DD; d += 256) {
    float e = emb[(size_t)tok * DD + d] * 32.0f;   // sqrt(1024)
    float expo = (float)(2 * (d >> 1)) * (1.0f / (float)DD);
    float ang = (float)s * __powf(10000.0f, -expo);
    float pe = (d & 1) ? __cosf(ang) : __sinf(ang);
    x[(size_t)bs * DD + d] = e + pe;
  }
}

// ---------------------------------------------------------------------------
// LayerNorm: one block per row of D=1024; optional f16 output
// ---------------------------------------------------------------------------
template <bool F16OUT>
__global__ __launch_bounds__(256) void layernorm_k(const float* __restrict__ x,
                                                   const float* __restrict__ g,
                                                   const float* __restrict__ bta,
                                                   void* __restrict__ out) {
  const int row = blockIdx.x, tid = threadIdx.x;
  const float4 v = *(const float4*)(x + (size_t)row * DD + tid * 4);
  float s  = v.x + v.y + v.z + v.w;
  float ss = v.x * v.x + v.y * v.y + v.z * v.z + v.w * v.w;
  #pragma unroll
  for (int m = 16; m >= 1; m >>= 1) {
    s  += __shfl_xor(s, m);
    ss += __shfl_xor(ss, m);
  }
  __shared__ float shm[8], shs[8];
  if ((tid & 31) == 0) { shm[tid >> 5] = s; shs[tid >> 5] = ss; }
  __syncthreads();
  float tot = 0.f, tots = 0.f;
  #pragma unroll
  for (int i = 0; i < 8; ++i) { tot += shm[i]; tots += shs[i]; }
  const float mean = tot * (1.0f / (float)DD);
  const float var  = tots * (1.0f / (float)DD) - mean * mean;
  const float rstd = rsqrtf(var + 1e-5f);
  const float in4[4] = {v.x, v.y, v.z, v.w};
  #pragma unroll
  for (int i = 0; i < 4; ++i) {
    const int d = tid * 4 + i;
    const float y = (in4[i] - mean) * rstd * g[d] + bta[d];
    if (F16OUT) ((_Float16*)out)[(size_t)row * DD + d] = (_Float16)y;
    else        ((float*)out)[(size_t)row * DD + d] = y;
  }
}

// ---------------------------------------------------------------------------
// WMMA GEMM: C[M,N] = act(A_f16[M,K] @ B_f32[K,N] + bias) (+ resid)
// Block 256 thr = 8 waves (2x4), block tile 128x256x32; per-wave 64x64.
// 16 WMMAs per wave per k-step; next k-tile prefetched (global_prefetch_b8).
// ---------------------------------------------------------------------------
template <bool RELU, bool RESID, bool OUTF16>
__global__ __launch_bounds__(256) void gemm_wmma(
    const _Float16* __restrict__ A, const float* __restrict__ B,
    const float* __restrict__ bias, const float* __restrict__ resid,
    void* __restrict__ out, int M, int N, int K) {
  __shared__ _Float16 As[128][32];   // 8 KB
  __shared__ _Float16 Bt[256][32];   // 16 KB, transposed: Bt[n][k]

  const int tid = threadIdx.x;
  const int lane = tid & 31, wave = tid >> 5;
  const int m0 = blockIdx.y * 128, n0 = blockIdx.x * 256;
  const int wm = (wave >> 2) * 64;   // 2 waves along M
  const int wn = (wave & 3) * 64;    // 4 waves along N
  const int lm = lane & 15, half = lane >> 4;

  v8f acc[4][4];
  #pragma unroll
  for (int i = 0; i < 4; ++i)
    #pragma unroll
    for (int j = 0; j < 4; ++j)
      #pragma unroll
      for (int r = 0; r < 8; ++r) acc[i][j][r] = 0.0f;

  const int ar = tid >> 1, aseg = tid & 1;     // A tile: 2 thr/row, 16 halves
  const int bn = tid;                          // B tile: 1 thr/col, 32 k's

  for (int k0 = 0; k0 < K; k0 += 32) {
    {  // A tile: 128x32 f16 (b128 x2 per thread)
      const uint4* src = (const uint4*)(A + (size_t)(m0 + ar) * K + k0 + aseg * 16);
      uint4 a0 = src[0], a1 = src[1];
      *(uint4*)&As[ar][aseg * 16]     = a0;
      *(uint4*)&As[ar][aseg * 16 + 8] = a1;
    }
    {  // B tile: 32x256 f32 -> f16, transposed into Bt[n][k]
      __align__(16) _Float16 tmp[32];
      #pragma unroll
      for (int i = 0; i < 32; ++i)
        tmp[i] = (_Float16)B[(size_t)(k0 + i) * N + (n0 + bn)];
      #pragma unroll
      for (int i = 0; i < 4; ++i)
        *(h8*)&Bt[bn][i * 8] = *(h8*)&tmp[i * 8];
    }
    if (k0 + 32 < K) {   // prefetch next k-tile into cache (no LOADcnt)
      __builtin_prefetch(A + (size_t)(m0 + ar) * K + k0 + 32 + aseg * 16, 0, 3);
      __builtin_prefetch(B + (size_t)(k0 + 32) * N + n0 + bn, 0, 3);
      __builtin_prefetch(B + (size_t)(k0 + 48) * N + n0 + bn, 0, 3);
    }
    __syncthreads();

    v16h af[4], bf[4];
    #pragma unroll
    for (int i = 0; i < 4; ++i)
      af[i] = ld_frag_a(&As[0][0], 32, wm + i * 16 + lm, 0, half);
    #pragma unroll
    for (int j = 0; j < 4; ++j)
      bf[j] = ld_frag_b(&Bt[0][0], 32, wn + j * 16 + lm, 0, half);

    #pragma unroll
    for (int i = 0; i < 4; ++i)
      #pragma unroll
      for (int j = 0; j < 4; ++j)
        acc[i][j] = wmma16(af[i], bf[j], acc[i][j]);
    __syncthreads();
  }

  // epilogue: C layout row = vgpr + 8*half, col = lane&15
  #pragma unroll
  for (int i = 0; i < 4; ++i)
    #pragma unroll
    for (int j = 0; j < 4; ++j) {
      const int col = n0 + wn + j * 16 + lm;
      #pragma unroll
      for (int r = 0; r < 8; ++r) {
        const int row = m0 + wm + i * 16 + r + half * 8;
        float val = acc[i][j][r] + bias[col];
        if (RELU)  val = fmaxf(val, 0.0f);
        if (RESID) val += resid[(size_t)row * N + col];
        if (OUTF16) ((_Float16*)out)[(size_t)row * N + col] = (_Float16)val;
        else        ((float*)out)[(size_t)row * N + col] = val;
      }
    }
}

// ---------------------------------------------------------------------------
// Fused flash attention: one block (4 waves) per (b,h,64-row q block).
// K head tile and V^T head tile resident in LDS (128 KB of 320 KB WGP LDS).
// Online softmax, P re-enters WMMA A layout via per-wave LDS scratch.
// ---------------------------------------------------------------------------
__global__ __launch_bounds__(128) void attn_k(const _Float16* __restrict__ q,
                                              const _Float16* __restrict__ kv,
                                              const int* __restrict__ lengths,
                                              _Float16* __restrict__ o) {
  __shared__ _Float16 Ksh[SS][DHEAD];      // 64 KB  [key][d]
  __shared__ _Float16 Vsh[DHEAD][SS];      // 64 KB  [d][key] (transposed)
  __shared__ _Float16 Qsh[4][16][DHEAD];   // 8 KB
  __shared__ _Float16 Psh[4][16][32];      // 4 KB

  const int bh = blockIdx.y, b = bh >> 4, h = bh & 15;
  const int qblk = blockIdx.x;             // 0..7
  const int tid = threadIdx.x, lane = tid & 31, w = tid >> 5;
  const int lm = lane & 15, half = lane >> 4;
  const int len = lengths[b];
  const float scale = 0.125f;              // 1/sqrt(64)

  // K head slice -> LDS
  for (int idx = tid; idx < SS * 8; idx += 128) {
    const int s = idx >> 3, seg = idx & 7;
    *(uint4*)&Ksh[s][seg * 8] =
        *(const uint4*)(kv + ((size_t)(b * SS + s)) * 2048 + h * 64 + seg * 8);
  }
  // V head slice -> LDS transposed
  for (int idx = tid; idx < SS * 8; idx += 128) {
    const int s = idx >> 3, seg = idx & 7;
    h8 v = *(const h8*)(kv + ((size_t)(b * SS + s)) * 2048 + 1024 + h * 64 + seg * 8);
    #pragma unroll
    for (int j = 0; j < 8; ++j) Vsh[seg * 8 + j][s] = v[j];
  }
  // Q tile (16 rows) per wave
  const int row0 = qblk * 64 + w * 16;
  for (int i = lane; i < 16 * 8; i += 32) {
    const int r = i >> 3, seg = i & 7;
    *(uint4*)&Qsh[w][r][seg * 8] =
        *(const uint4*)(q + ((size_t)(b * SS + row0 + r)) * DD + h * 64 + seg * 8);
  }
  __syncthreads();

  const v16h qa0 = ld_frag_a(&Qsh[w][0][0], DHEAD, lm, 0, half);
  const v16h qa1 = ld_frag_a(&Qsh[w][0][0], DHEAD, lm, 32, half);

  float mrun[8], lrun[8];
  v8f oacc[4];
  #pragma unroll
  for (int j = 0; j < 8; ++j) { mrun[j] = -1e30f; lrun[j] = 0.0f; }
  #pragma unroll
  for (int nc = 0; nc < 4; ++nc)
    #pragma unroll
    for (int j = 0; j < 8; ++j) oacc[nc][j] = 0.0f;

  for (int kb = 0; kb < SS; kb += 32) {
    v8f s0, s1;
    #pragma unroll
    for (int j = 0; j < 8; ++j) { s0[j] = 0.0f; s1[j] = 0.0f; }
    // scores: q(16x64) @ K^T(64x32) -> two 16x16 tiles
    s0 = wmma16(qa0, ld_frag_b(&Ksh[0][0], DHEAD, kb + lm,       0, half), s0);
    s0 = wmma16(qa1, ld_frag_b(&Ksh[0][0], DHEAD, kb + lm,      32, half), s0);
    s1 = wmma16(qa0, ld_frag_b(&Ksh[0][0], DHEAD, kb + 16 + lm,  0, half), s1);
    s1 = wmma16(qa1, ld_frag_b(&Ksh[0][0], DHEAD, kb + 16 + lm, 32, half), s1);

    const bool v0 = (kb + lm) < len;
    const bool v1 = (kb + 16 + lm) < len;
    float al[8];
    #pragma unroll
    for (int j = 0; j < 8; ++j) {
      float a = v0 ? s0[j] * scale : -1e30f;
      float c = v1 ? s1[j] * scale : -1e30f;
      float mx = fmaxf(a, c);
      #pragma unroll
      for (int m = 1; m <= 8; m <<= 1) mx = fmaxf(mx, __shfl_xor(mx, m));
      const float mnew  = fmaxf(mrun[j], mx);
      const float alpha = __expf(mrun[j] - mnew);
      const float p0 = __expf(a - mnew);
      const float p1 = __expf(c - mnew);
      float rs = p0 + p1;
      #pragma unroll
      for (int m = 1; m <= 8; m <<= 1) rs += __shfl_xor(rs, m);
      lrun[j] = lrun[j] * alpha + rs;
      mrun[j] = mnew;
      al[j] = alpha;
      Psh[w][j + half * 8][lm]      = (_Float16)p0;   // C layout -> LDS
      Psh[w][j + half * 8][lm + 16] = (_Float16)p1;
    }
    __asm__ volatile("s_wait_dscnt 0" ::: "memory");  // wave-local RAW on Psh
    const v16h pa = ld_frag_a(&Psh[w][0][0], 32, lm, 0, half);
    #pragma unroll
    for (int nc = 0; nc < 4; ++nc) {
      #pragma unroll
      for (int j = 0; j < 8; ++j) oacc[nc][j] *= al[j];
      oacc[nc] = wmma16(pa, ld_frag_b(&Vsh[0][0], SS, nc * 16 + lm, kb, half),
                        oacc[nc]);
    }
  }

  #pragma unroll
  for (int j = 0; j < 8; ++j) {
    const float inv = 1.0f / lrun[j];
    const int row = b * SS + row0 + j + half * 8;
    #pragma unroll
    for (int nc = 0; nc < 4; ++nc)
      o[(size_t)row * DD + h * 64 + nc * 16 + lm] =
          (_Float16)(oacc[nc][j] * inv);
  }
}

// ---------------------------------------------------------------------------
// Host orchestration
// ---------------------------------------------------------------------------
extern "C" void kernel_launch(void* const* d_in, const int* in_sizes, int n_in,
                              void* d_out, int out_size, void* d_ws,
                              size_t ws_size, hipStream_t stream) {
  (void)in_sizes; (void)n_in; (void)out_size; (void)ws_size;
  const int*   tokens  = (const int*)d_in[0];
  const int*   lengths = (const int*)d_in[1];
  const float* emb     = (const float*)d_in[2];
  const float* Wq      = (const float*)d_in[3];
  const float* bq      = (const float*)d_in[4];
  const float* Wkv     = (const float*)d_in[5];
  const float* bkv     = (const float*)d_in[6];
  const float* Wo      = (const float*)d_in[7];
  const float* bo      = (const float*)d_in[8];
  const float* lag     = (const float*)d_in[9];
  const float* lab     = (const float*)d_in[10];
  const float* W1      = (const float*)d_in[11];
  const float* b1      = (const float*)d_in[12];
  const float* W2      = (const float*)d_in[13];
  const float* b2      = (const float*)d_in[14];
  const float* lfg     = (const float*)d_in[15];
  const float* lfb     = (const float*)d_in[16];
  const float* lng     = (const float*)d_in[17];
  const float* lnb     = (const float*)d_in[18];

  // workspace layout (~88 MB)
  char* w = (char*)d_ws;
  float*     x    = (float*)w;      w += (size_t)NTOK * DD * 4;   // 16 MB
  _Float16*  h16  = (_Float16*)w;   w += (size_t)NTOK * DD * 2;   //  8 MB
  _Float16*  q16  = (_Float16*)w;   w += (size_t)NTOK * DD * 2;   //  8 MB
  _Float16*  kv16 = (_Float16*)w;   w += (size_t)NTOK * 2048 * 2; // 16 MB
  _Float16*  o16  = (_Float16*)w;   w += (size_t)NTOK * DD * 2;   //  8 MB
  _Float16*  f116 = (_Float16*)w;   w += (size_t)NTOK * DI * 2;   // 32 MB

  embed_k<<<NTOK, 256, 0, stream>>>(tokens, emb, x);

  for (int l = 0; l < NL; ++l) {
    const float* wq  = Wq  + (size_t)l * DD * DD;
    const float* wkv = Wkv + (size_t)l * DD * 2048;
    const float* wo  = Wo  + (size_t)l * DD * DD;
    const float* w1  = W1  + (size_t)l * DD * DI;
    const float* w2  = W2  + (size_t)l * DI * DD;

    // attention sublayer
    layernorm_k<true><<<NTOK, 256, 0, stream>>>(x, lag + (size_t)l * DD,
                                                lab + (size_t)l * DD, h16);
    gemm_wmma<false, false, true><<<dim3(DD / 256, NTOK / 128), 256, 0, stream>>>(
        h16, wq, bq + (size_t)l * DD, nullptr, q16, NTOK, DD, DD);
    gemm_wmma<false, false, true><<<dim3(2048 / 256, NTOK / 128), 256, 0, stream>>>(
        h16, wkv, bkv + (size_t)l * 2048, nullptr, kv16, NTOK, 2048, DD);
    attn_k<<<dim3(SS / 64, BB * HH), 128, 0, stream>>>(q16, kv16, lengths, o16);
    gemm_wmma<false, true, false><<<dim3(DD / 256, NTOK / 128), 256, 0, stream>>>(
        o16, wo, bo + (size_t)l * DD, x, x, NTOK, DD, DD);

    // FFN sublayer
    layernorm_k<true><<<NTOK, 256, 0, stream>>>(x, lfg + (size_t)l * DD,
                                                lfb + (size_t)l * DD, h16);
    gemm_wmma<true, false, true><<<dim3(DI / 256, NTOK / 128), 256, 0, stream>>>(
        h16, w1, b1 + (size_t)l * DI, nullptr, f116, NTOK, DI, DD);
    gemm_wmma<false, true, false><<<dim3(DD / 256, NTOK / 128), 256, 0, stream>>>(
        f116, w2, b2 + (size_t)l * DD, x, x, NTOK, DD, DI);
  }

  layernorm_k<false><<<NTOK, 256, 0, stream>>>(x, lng, lnb, d_out);
}
